// CascadeInpaintingNet_87290915323962
// MI455X (gfx1250) — matrix operand order repaired and checked by reading, not compiled, not run on hardware
//
#include <hip/hip_runtime.h>
#include <hip/hip_bf16.h>

// ---------------------------------------------------------------------------
// CascadeInpaintingNet on MI455X (gfx1250, wave32, WMMA, async-to-LDS)
//
// Pipeline (per image n of 8):
//   k1: conv3x3 1->64 + lrelu        (VALU)                         -> h1 (half, ch-last)
//   k2: conv3x3 64->64 + lrelu       (WMMA f32_16x16x32_f16)        -> h2 (half, ch-last)
//   k3: conv1x1 64->9 + softmax + 3x upsample-multiply              -> out (f32) + outh (half)
//   k4: conv3x3 1->32 + lrelu        (VALU)                         -> hr1 (half, ch-last)
//   k5: conv3x3 32->32 + lrelu       (WMMA f32_16x16x32_f16)        -> hr2 (half, ch-last)
//   k6: conv3x3 32->1 + residual add + clip                         -> d_out
//
// 3x3 convs = 9 accumulated per-tap GEMMs; channel-last layout makes every
// B-fragment two runs of 8 contiguous halves in LDS (verified: merges to ds_load_b128).
// GEMM halo staging uses GLOBAL_LOAD_ASYNC_TO_LDS_B128 (ASYNCcnt-tracked,
// VGPR-bypassing); A fragments are loaded once per tap and feed two
// back-to-back WMMAs (dual pixel-subtile accumulators).
// ---------------------------------------------------------------------------

typedef _Float16 half_t;
typedef __attribute__((ext_vector_type(16))) half_t v16h;
typedef __attribute__((ext_vector_type(8)))  float  v8f;
typedef int v4i_t __attribute__((vector_size(16)));   // matches builtin param type

#define IMG_H 256
#define IMG_W 256
#define UP_H  768
#define UP_W  768
#define HID   64
#define RF    32
#define NPIX  (IMG_H * IMG_W)      // 65536
#define UPIX  (UP_H * UP_W)        // 589824

#define AS_GLOBAL __attribute__((address_space(1)))
#define AS_LDS    __attribute__((address_space(3)))

// A-matrix 16x32 f16 VGPR layout (ISA 7.12.2): lane L -> M = L%16, g = L/16;
// VGPR v holds K pair { Kbase(v,g), Kbase(v,g)+1 }.
__device__ __host__ __forceinline__ int a_kbase(int v, int g) {
  return (v < 4 ? 2 * v : 16 + 2 * (v - 4)) + 8 * g;
}
// B-matrix 32x16 f16 layout: lane L -> N = L%16, g = L/16; VGPR v holds
// K pair { Kbase(v,g), Kbase(v,g)+1 } (same K order as the A fragment).
__device__ __forceinline__ int b_kbase(int v, int g) {
  return (v < 4 ? 2 * v : 16 + 2 * (v - 4)) + 8 * g;
}

__device__ __forceinline__ float lrelu(float x) {
  return x > 0.f ? x : 0.2f * x;
}

// 16-byte global -> LDS copy on the gfx1250 async path when available.
__device__ __forceinline__ void stage16(const uint4* __restrict__ g, uint4* l) {
#if defined(__has_builtin) && __has_builtin(__builtin_amdgcn_global_load_async_to_lds_b128)
  __builtin_amdgcn_global_load_async_to_lds_b128(
      (AS_GLOBAL v4i_t*)(uintptr_t)g, (AS_LDS v4i_t*)l, 0, 0);
#else
  *l = *g;
#endif
}

__device__ __forceinline__ void wait_async_lds() {
#if defined(__has_builtin) && __has_builtin(__builtin_amdgcn_s_wait_asynccnt)
  __builtin_amdgcn_s_wait_asynccnt(0);
#elif defined(__has_builtin) && __has_builtin(__builtin_amdgcn_global_load_async_to_lds_b128)
  asm volatile("s_wait_asynccnt 0x0" ::: "memory");
#endif
}

// ---------------------------------------------------------------------------
// Kernel 0: pack wg2 (64x64x3x3) and r2 (32x32x3x3) weights into WMMA
// A-fragment order so GEMM waves load them as contiguous 32B chunks.
// packA2: [tap 9][kc 2][mtile 4][lane 32][16 halves]
// packA5: [tap 9][mtile 2][lane 32][16 halves]
// ---------------------------------------------------------------------------
__global__ void pack_weights(const float* __restrict__ wg2_w,
                             const float* __restrict__ r2_w,
                             half_t* __restrict__ packA2,
                             half_t* __restrict__ packA5) {
  int idx = blockIdx.x * blockDim.x + threadIdx.x;
  const int N2 = 9 * 2 * 4 * 32 * 16;   // 36864
  const int N5 = 9 * 2 * 32 * 16;       // 9216
  if (idx < N2) {
    int hh = idx & 1;
    int v  = (idx >> 1) & 7;
    int L  = (idx >> 4) & 31;
    int mt = (idx >> 9) & 3;
    int kc = (idx >> 11) & 1;
    int t  = idx >> 12;                 // 0..8
    int M  = mt * 16 + (L & 15);
    int K  = kc * 32 + a_kbase(v, L >> 4) + hh;
    packA2[idx] = (half_t)wg2_w[(M * 64 + K) * 9 + t];
  } else if (idx < N2 + N5) {
    int i5 = idx - N2;
    int hh = i5 & 1;
    int v  = (i5 >> 1) & 7;
    int L  = (i5 >> 4) & 31;
    int mt = (i5 >> 9) & 1;
    int t  = i5 >> 10;                  // 0..8
    int M  = mt * 16 + (L & 15);
    int K  = a_kbase(v, L >> 4) + hh;
    packA5[i5] = (half_t)r2_w[(M * 32 + K) * 9 + t];
  }
}

// ---------------------------------------------------------------------------
// Kernel 1: wg1 conv3x3 (1 -> 64) + lrelu, fp32 in -> half channel-last out
// ---------------------------------------------------------------------------
__global__ void k1_wg1(const float* __restrict__ x,
                       const float* __restrict__ w,
                       const float* __restrict__ b,
                       half_t* __restrict__ h1, int n) {
  int p = blockIdx.x * blockDim.x + threadIdx.x;
  if (p >= NPIX) return;
  int y = p >> 8, xx = p & 255;
  const float* xi = x + (size_t)n * NPIX;
  float v[9];
#pragma unroll
  for (int dy = 0; dy < 3; ++dy)
#pragma unroll
    for (int dx = 0; dx < 3; ++dx) {
      int gy = y + dy - 1, gx = xx + dx - 1;
      v[dy * 3 + dx] =
          (gy >= 0 && gy < IMG_H && gx >= 0 && gx < IMG_W) ? xi[gy * IMG_W + gx] : 0.f;
    }
  alignas(16) half_t out[HID];
#pragma unroll 8
  for (int c = 0; c < HID; ++c) {
    float acc = b[c];
#pragma unroll
    for (int t = 0; t < 9; ++t) acc += w[c * 9 + t] * v[t];
    out[c] = (half_t)lrelu(acc);
  }
  uint4* dst = reinterpret_cast<uint4*>(h1 + (size_t)p * HID);
  const uint4* src = reinterpret_cast<const uint4*>(out);
#pragma unroll
  for (int i = 0; i < 8; ++i) dst[i] = src[i];
}

// ---------------------------------------------------------------------------
// Kernel 2: wg2 conv3x3 (64 -> 64) + lrelu via WMMA.
// Block = 128 thr (4 waves), tile = 64 out-ch x 32 pixels of one row.
// Per wave: 18 A fragments, each feeding 2 WMMAs (dual subtile accumulators).
// ---------------------------------------------------------------------------
__global__ void __launch_bounds__(128) k2_wg2(const half_t* __restrict__ h1,
                                              const half_t* __restrict__ packA,
                                              const float* __restrict__ bias,
                                              half_t* __restrict__ h2) {
  __shared__ alignas(16) half_t lds[3 * 34 * HID];   // 13056 B halo tile
  int x0 = blockIdx.x * 32;
  int y  = blockIdx.y;
  int tid = threadIdx.x;

  // Stage 3 rows x 34 cols x 64ch halo into LDS (async path; zeros via ds).
  for (int pi = tid; pi < 3 * 34; pi += 128) {
    int r = pi / 34, c = pi % 34;
    int gy = y + r - 1, gx = x0 + c - 1;
    uint4* dst = reinterpret_cast<uint4*>(&lds[pi * HID]);
    if (gy >= 0 && gy < IMG_H && gx >= 0 && gx < IMG_W) {
      const uint4* src =
          reinterpret_cast<const uint4*>(h1 + ((size_t)gy * IMG_W + gx) * HID);
#pragma unroll
      for (int i = 0; i < 8; ++i) stage16(src + i, dst + i);
    } else {
      uint4 z{0, 0, 0, 0};
#pragma unroll
      for (int i = 0; i < 8; ++i) dst[i] = z;
    }
  }
  wait_async_lds();
  __syncthreads();

  int wave = tid >> 5;
  int lane = tid & 31;
  int ln = lane & 15, g = lane >> 4;
  int m0 = wave * 16;

  v8f acc0, acc1;
#pragma unroll
  for (int v = 0; v < 8; ++v) {
    float bv = bias[m0 + v + 8 * g];
    acc0[v] = bv;
    acc1[v] = bv;
  }

#pragma unroll
  for (int t = 0; t < 9; ++t) {
    int dy = t / 3, dx = t % 3;
#pragma unroll
    for (int kc = 0; kc < 2; ++kc) {
      v16h a = *reinterpret_cast<const v16h*>(
          packA + (((t * 2 + kc) * 4 + wave) * 32 + lane) * 16);
      const half_t* b0 = &lds[(dy * 34 + ln + dx) * HID];
      const half_t* b1 = &lds[(dy * 34 + 16 + ln + dx) * HID];
      v16h bf0, bf1;
#pragma unroll
      for (int v = 0; v < 8; ++v) {
        int kb = kc * 32 + b_kbase(v, g);
        bf0[2 * v]     = b0[kb];
        bf0[2 * v + 1] = b0[kb + 1];
        bf1[2 * v]     = b1[kb];
        bf1[2 * v + 1] = b1[kb + 1];
      }
      acc0 = __builtin_amdgcn_wmma_f32_16x16x32_f16(
          false, a, false, bf0, (short)0, acc0, false, false);
      acc1 = __builtin_amdgcn_wmma_f32_16x16x32_f16(
          false, a, false, bf1, (short)0, acc1, false, false);
    }
  }

#pragma unroll
  for (int s = 0; s < 2; ++s) {
    const v8f& acc = s ? acc1 : acc0;
    alignas(16) half_t hv[8];
#pragma unroll
    for (int v = 0; v < 8; ++v) hv[v] = (half_t)lrelu(acc[v]);
    int px = x0 + s * 16 + ln;
    half_t* dst = h2 + ((size_t)y * IMG_W + px) * HID + m0 + 8 * g;
    *reinterpret_cast<uint4*>(dst) = *reinterpret_cast<const uint4*>(hv);
  }
}

// ---------------------------------------------------------------------------
// Kernel 3: wg3 1x1 (64 -> 9) + softmax + per-pixel 3x3 kernel expansion.
// ---------------------------------------------------------------------------
__global__ void k3_wg3(const half_t* __restrict__ h2,
                       const float* __restrict__ w,
                       const float* __restrict__ b,
                       const float* __restrict__ x,
                       float* __restrict__ outf,
                       half_t* __restrict__ outh, int n) {
  int p = blockIdx.x * blockDim.x + threadIdx.x;
  if (p >= NPIX) return;
  int y = p >> 8, xx = p & 255;
  alignas(16) half_t hv[HID];
  const uint4* src = reinterpret_cast<const uint4*>(h2 + (size_t)p * HID);
  uint4* dv = reinterpret_cast<uint4*>(hv);
#pragma unroll
  for (int i = 0; i < 8; ++i) dv[i] = src[i];

  float logits[9];
  float mx = -1e30f;
#pragma unroll
  for (int k = 0; k < 9; ++k) {
    float acc = b[k];
#pragma unroll
    for (int c = 0; c < HID; ++c) acc += w[k * HID + c] * (float)hv[c];
    logits[k] = acc;
    mx = fmaxf(mx, acc);
  }
  float se = 0.f;
#pragma unroll
  for (int k = 0; k < 9; ++k) {
    logits[k] = __expf(logits[k] - mx);
    se += logits[k];
  }
  float inv = 1.f / se;
  float xv = x[(size_t)n * NPIX + p];
#pragma unroll
  for (int ky = 0; ky < 3; ++ky)
#pragma unroll
    for (int kx = 0; kx < 3; ++kx) {
      float o = xv * logits[ky * 3 + kx] * inv;
      size_t op = (size_t)(3 * y + ky) * UP_W + (3 * xx + kx);
      outf[(size_t)n * UPIX + op] = o;
      outh[op] = (half_t)o;
    }
}

// ---------------------------------------------------------------------------
// Kernel 4: r1 conv3x3 (1 -> 32) + lrelu on the 768x768 image.
// ---------------------------------------------------------------------------
__global__ void k4_r1(const half_t* __restrict__ outh,
                      const float* __restrict__ w,
                      const float* __restrict__ b,
                      half_t* __restrict__ hr1) {
  int p = blockIdx.x * blockDim.x + threadIdx.x;
  if (p >= UPIX) return;
  int y = p / UP_W, xx = p - y * UP_W;
  float v[9];
#pragma unroll
  for (int dy = 0; dy < 3; ++dy)
#pragma unroll
    for (int dx = 0; dx < 3; ++dx) {
      int gy = y + dy - 1, gx = xx + dx - 1;
      v[dy * 3 + dx] = (gy >= 0 && gy < UP_H && gx >= 0 && gx < UP_W)
                           ? (float)outh[(size_t)gy * UP_W + gx]
                           : 0.f;
    }
  alignas(16) half_t out[RF];
#pragma unroll 8
  for (int c = 0; c < RF; ++c) {
    float acc = b[c];
#pragma unroll
    for (int t = 0; t < 9; ++t) acc += w[c * 9 + t] * v[t];
    out[c] = (half_t)lrelu(acc);
  }
  uint4* dst = reinterpret_cast<uint4*>(hr1 + (size_t)p * RF);
  const uint4* src = reinterpret_cast<const uint4*>(out);
#pragma unroll
  for (int i = 0; i < 4; ++i) dst[i] = src[i];
}

// ---------------------------------------------------------------------------
// Kernel 5: r2 conv3x3 (32 -> 32) + lrelu via WMMA.
// Block = 64 thr (2 waves), tile = 32 out-ch x 32 pixels. 9 A fragments,
// each feeding 2 WMMAs (dual subtile accumulators).
// ---------------------------------------------------------------------------
__global__ void __launch_bounds__(64) k5_r2(const half_t* __restrict__ hr1,
                                            const half_t* __restrict__ packA,
                                            const float* __restrict__ bias,
                                            half_t* __restrict__ hr2) {
  __shared__ alignas(16) half_t lds[3 * 34 * RF];    // 6528 B halo tile
  int x0 = blockIdx.x * 32;
  int y  = blockIdx.y;
  int tid = threadIdx.x;

  for (int pi = tid; pi < 3 * 34; pi += 64) {
    int r = pi / 34, c = pi % 34;
    int gy = y + r - 1, gx = x0 + c - 1;
    uint4* dst = reinterpret_cast<uint4*>(&lds[pi * RF]);
    if (gy >= 0 && gy < UP_H && gx >= 0 && gx < UP_W) {
      const uint4* src =
          reinterpret_cast<const uint4*>(hr1 + ((size_t)gy * UP_W + gx) * RF);
#pragma unroll
      for (int i = 0; i < 4; ++i) stage16(src + i, dst + i);
    } else {
      uint4 z{0, 0, 0, 0};
#pragma unroll
      for (int i = 0; i < 4; ++i) dst[i] = z;
    }
  }
  wait_async_lds();
  __syncthreads();

  int wave = tid >> 5;
  int lane = tid & 31;
  int ln = lane & 15, g = lane >> 4;
  int m0 = wave * 16;

  v8f acc0, acc1;
#pragma unroll
  for (int v = 0; v < 8; ++v) {
    float bv = bias[m0 + v + 8 * g];
    acc0[v] = bv;
    acc1[v] = bv;
  }

#pragma unroll
  for (int t = 0; t < 9; ++t) {
    int dy = t / 3, dx = t % 3;
    v16h a = *reinterpret_cast<const v16h*>(
        packA + (((t * 2 + wave) * 32) + lane) * 16);
    const half_t* b0 = &lds[(dy * 34 + ln + dx) * RF];
    const half_t* b1 = &lds[(dy * 34 + 16 + ln + dx) * RF];
    v16h bf0, bf1;
#pragma unroll
    for (int v = 0; v < 8; ++v) {
      int kb = b_kbase(v, g);
      bf0[2 * v]     = b0[kb];
      bf0[2 * v + 1] = b0[kb + 1];
      bf1[2 * v]     = b1[kb];
      bf1[2 * v + 1] = b1[kb + 1];
    }
    acc0 = __builtin_amdgcn_wmma_f32_16x16x32_f16(
        false, a, false, bf0, (short)0, acc0, false, false);
    acc1 = __builtin_amdgcn_wmma_f32_16x16x32_f16(
        false, a, false, bf1, (short)0, acc1, false, false);
  }

#pragma unroll
  for (int s = 0; s < 2; ++s) {
    const v8f& acc = s ? acc1 : acc0;
    alignas(16) half_t hv[8];
#pragma unroll
    for (int v = 0; v < 8; ++v) hv[v] = (half_t)lrelu(acc[v]);
    int px = x0 + s * 16 + ln;
    half_t* dst = hr2 + ((size_t)y * UP_W + px) * RF + m0 + 8 * g;
    *reinterpret_cast<uint4*>(dst) = *reinterpret_cast<const uint4*>(hv);
  }
}

// ---------------------------------------------------------------------------
// Kernel 6: r3 conv3x3 (32 -> 1) + residual add + clip, in-place on d_out.
// ---------------------------------------------------------------------------
__global__ void k6_r3(const half_t* __restrict__ hr2,
                      const float* __restrict__ w,
                      const float* __restrict__ b,
                      float* __restrict__ io, int n) {
  int p = blockIdx.x * blockDim.x + threadIdx.x;
  if (p >= UPIX) return;
  int y = p / UP_W, xx = p - y * UP_W;
  float acc = b[0];
#pragma unroll
  for (int t = 0; t < 9; ++t) {
    int dy = t / 3, dx = t % 3;
    int gy = y + dy - 1, gx = xx + dx - 1;
    if (gy >= 0 && gy < UP_H && gx >= 0 && gx < UP_W) {
      alignas(16) half_t pv[RF];
      const uint4* s4 =
          reinterpret_cast<const uint4*>(hr2 + ((size_t)gy * UP_W + gx) * RF);
      uint4* d4 = reinterpret_cast<uint4*>(pv);
#pragma unroll
      for (int i = 0; i < 4; ++i) d4[i] = s4[i];
#pragma unroll
      for (int c = 0; c < RF; ++c) acc += w[c * 9 + t] * (float)pv[c];
    }
  }
  size_t op = (size_t)n * UPIX + p;
  float o = io[op] + acc;
  io[op] = fminf(fmaxf(o, 0.f), 1.f);
}

// ---------------------------------------------------------------------------
// Host launcher. Batch is looped so workspace stays ~94 MB.
// ---------------------------------------------------------------------------
extern "C" void kernel_launch(void* const* d_in, const int* in_sizes, int n_in,
                              void* d_out, int out_size, void* d_ws, size_t ws_size,
                              hipStream_t stream) {
  const float* x    = (const float*)d_in[0];
  const float* wg1w = (const float*)d_in[1];
  const float* wg1b = (const float*)d_in[2];
  const float* wg2w = (const float*)d_in[3];
  const float* wg2b = (const float*)d_in[4];
  const float* wg3w = (const float*)d_in[5];
  const float* wg3b = (const float*)d_in[6];
  const float* r1w  = (const float*)d_in[7];
  const float* r1b  = (const float*)d_in[8];
  const float* r2w  = (const float*)d_in[9];
  const float* r2b  = (const float*)d_in[10];
  const float* r3w  = (const float*)d_in[11];
  const float* r3b  = (const float*)d_in[12];
  float* out = (float*)d_out;
  char* ws = (char*)d_ws;

  // Workspace layout (bytes), all 256B-aligned.
  size_t off = 0;
  half_t* packA2 = (half_t*)(ws + off); off += 9 * 2 * 4 * 32 * 16 * sizeof(half_t); // 73728
  half_t* packA5 = (half_t*)(ws + off); off += 9 * 2 * 32 * 16 * sizeof(half_t);     // 18432
  off = (off + 255) & ~(size_t)255;
  half_t* h1   = (half_t*)(ws + off); off += (size_t)NPIX * HID * sizeof(half_t);    // 8 MiB
  half_t* h2   = (half_t*)(ws + off); off += (size_t)NPIX * HID * sizeof(half_t);    // 8 MiB
  half_t* outh = (half_t*)(ws + off); off += (size_t)UPIX * sizeof(half_t);          // 1.1 MiB
  off = (off + 255) & ~(size_t)255;
  half_t* hr1  = (half_t*)(ws + off); off += (size_t)UPIX * RF * sizeof(half_t);     // 36 MiB
  half_t* hr2  = (half_t*)(ws + off);                                                // 36 MiB

  pack_weights<<<(9 * 2 * 4 * 32 * 16 + 9 * 2 * 32 * 16 + 255) / 256, 256, 0, stream>>>(
      wg2w, r2w, packA2, packA5);

  for (int n = 0; n < 8; ++n) {
    k1_wg1<<<NPIX / 256, 256, 0, stream>>>(x, wg1w, wg1b, h1, n);
    k2_wg2<<<dim3(IMG_W / 32, IMG_H), 128, 0, stream>>>(h1, packA2, wg2b, h2);
    k3_wg3<<<NPIX / 256, 256, 0, stream>>>(h2, wg3w, wg3b, x, out, outh, n);
    k4_r1<<<UPIX / 256, 256, 0, stream>>>(outh, r1w, r1b, hr1);
    k5_r2<<<dim3(UP_W / 32, UP_H), 64, 0, stream>>>(hr1, packA5, r2b, hr2);
    k6_r3<<<UPIX / 256, 256, 0, stream>>>(hr2, r3w, r3b, out, n);
  }
}
